// Net_6992206758227
// MI455X (gfx1250) — matrix operand order repaired and checked by reading, not compiled, not run on hardware
//
#include <hip/hip_runtime.h>
#include <hip/hip_bf16.h>
#include <stdint.h>

// ---------------------------------------------------------------------------
// 3-layer GCN link predictor for MI455X (gfx1250, wave32, WMMA).
//  - GEMMs via v_wmma_f32_16x16x32_f16 (f32 accumulate), W staged in LDS as f16
//  - edge aggregation via L2-resident gather + global_atomic_add_f32 scatter
//  - everything (25.6MB tables) fits in the 192MB L2 -> scatter is L2-bound
// ---------------------------------------------------------------------------

typedef __attribute__((ext_vector_type(16))) _Float16 v16h;
typedef __attribute__((ext_vector_type(8)))  float    v8f;

#define DROP_P 0.3f
#define KDIM   128

// ------------------------- degree / norm ------------------------------------
__global__ void init_deg_kernel(float* deg, int n) {
  int i = blockIdx.x * blockDim.x + threadIdx.x;
  if (i < n) deg[i] = 1.0f;                       // self-loop contribution
}

__global__ void accum_deg_kernel(float* deg, const long long* __restrict__ dst,
                                 long long nE) {
  long long e = (long long)blockIdx.x * blockDim.x + threadIdx.x;
  if (e < nE) atomicAdd(&deg[(int)dst[e]], 1.0f);
}

__global__ void finish_dinv_kernel(float* deg, int n) {
  int i = blockIdx.x * blockDim.x + threadIdx.x;
  if (i < n) { float d = deg[i]; deg[i] = (d > 0.f) ? rsqrtf(d) : 0.f; }
}

// ------------------------- WMMA GEMM: H = X(Mx128) @ W(128xNOUT) ------------
// One wave computes a 16-row x NOUT strip. 8 waves/block -> 128 rows/block.
template <int NT>                                  // NOUT = NT*16 (8 or 4)
__global__ __launch_bounds__(256)
void gemm_f16_wmma_kernel(const float* __restrict__ X,
                          const float* __restrict__ W,
                          float* __restrict__ H, int M) {
  constexpr int NOUT = NT * 16;
  __shared__ __align__(16) _Float16 ldsW[NOUT * KDIM];   // [n][k]

  // cooperative f32 -> f16 transpose-convert of W (row-major [k][n]) into LDS
  for (int i = threadIdx.x; i < KDIM * NOUT; i += blockDim.x) {
    int k = i / NOUT, n = i % NOUT;
    ldsW[n * KDIM + k] = (_Float16)W[i];
  }
  __syncthreads();

  const int wave = threadIdx.x >> 5;
  const int lane = threadIdx.x & 31;
  const int col  = lane & 15;                      // N within tile / M row idx
  const int sel  = lane >> 4;                      // lane half (K-split)
  const int m0   = (blockIdx.x * 8 + wave) * 16;

  v8f acc[NT] = {};

  const int  m      = m0 + col;
  const bool mvalid = (m < M);
  const float* xrow = X + (size_t)(mvalid ? m : 0) * KDIM;

  for (int kb = 0; kb < KDIM; kb += 32) {
    // ---- A fragment (16x32, contiguous K per lane half, same order as B) ----
    // lanes 0-15: K = kb+0..15 ; lanes 16-31: K = kb+16..31 (contiguous, matches B)
    v16h a;
    {
      const float4* p0 = (const float4*)(xrow + kb + sel * 16);       // 2x b128
      const float4* p1 = (const float4*)(xrow + kb + sel * 16 + 8);
      float4 f0 = p0[0], f1 = p0[1], f2 = p1[0], f3 = p1[1];
      a[0]  = (_Float16)f0.x; a[1]  = (_Float16)f0.y; a[2]  = (_Float16)f0.z; a[3]  = (_Float16)f0.w;
      a[4]  = (_Float16)f1.x; a[5]  = (_Float16)f1.y; a[6]  = (_Float16)f1.z; a[7]  = (_Float16)f1.w;
      a[8]  = (_Float16)f2.x; a[9]  = (_Float16)f2.y; a[10] = (_Float16)f2.z; a[11] = (_Float16)f2.w;
      a[12] = (_Float16)f3.x; a[13] = (_Float16)f3.y; a[14] = (_Float16)f3.z; a[15] = (_Float16)f3.w;
    }
#pragma unroll
    for (int t = 0; t < NT; ++t) {
      // ---- B fragment: col n = t*16+col, K = kb + sel*16 + 0..15 (contig in LDS)
      const int n = t * 16 + col;
      union { v16h v; uint4 u[2]; } bf;
      const uint4* bp = (const uint4*)&ldsW[n * KDIM + kb + sel * 16];
      bf.u[0] = bp[0]; bf.u[1] = bp[1];
      acc[t] = __builtin_amdgcn_wmma_f32_16x16x32_f16(
          /*neg_a=*/false, a, /*neg_b=*/false, bf.v,
          /*c_mod=*/(short)0, acc[t], /*reuse_a=*/false, /*reuse_b=*/false);
    }
  }

  // ---- store C/D (VGPR r: lanes 0-15 -> M=r, lanes 16-31 -> M=r+8) ----
#pragma unroll
  for (int t = 0; t < NT; ++t) {
    const int n = t * 16 + col;
#pragma unroll
    for (int r = 0; r < 8; ++r) {
      int mm = m0 + r + sel * 8;
      if (mm < M) H[(size_t)mm * NOUT + n] = acc[t][r];
    }
  }
}

// ------------------------- bias init ----------------------------------------
__global__ void init_bias_kernel(float* out, const float* __restrict__ b,
                                 int cmask, long long total) {
  long long i = (long long)blockIdx.x * blockDim.x + threadIdx.x;
  if (i < total) out[i] = b[(int)(i & cmask)];
}

// ------------------------- edge scatter: out[dst] += h[src]*norm ------------
// One float4 per thread; 32 (C=128) or 16 (C=64) consecutive threads per edge
// -> coalesced b128 row reads, global_atomic_add_f32 scatter (L2-resident).
__global__ void scatter_edges_kernel(const float* __restrict__ H,
                                     float* __restrict__ OUT,
                                     const long long* __restrict__ src,
                                     const long long* __restrict__ dst,
                                     const float* __restrict__ dinv,
                                     long long nE, int nNodes, int sh, int C) {
  long long t = (long long)blockIdx.x * blockDim.x + threadIdx.x;
  long long total = (nE + nNodes) << sh;
  if (t >= total) return;
  long long e = t >> sh;
  int q = (int)(t & ((1 << sh) - 1));
  int s, d;
  if (e < nE) { s = (int)src[e]; d = (int)dst[e]; }
  else        { s = d = (int)(e - nE); }
  float nrm = dinv[s] * dinv[d];
  float4 hv = ((const float4*)(H + (size_t)s * C))[q];
  float* o = OUT + (size_t)d * C + q * 4;
  atomicAdd(o + 0, hv.x * nrm);
  atomicAdd(o + 1, hv.y * nrm);
  atomicAdd(o + 2, hv.z * nrm);
  atomicAdd(o + 3, hv.w * nrm);
}

// ------------------------- dropout + relu (deterministic hash RNG) ----------
__device__ __forceinline__ float hash_uniform(unsigned x) {
  x ^= x >> 17; x *= 0xed5ad4bbu;
  x ^= x >> 11; x *= 0xac4c1b51u;
  x ^= x >> 15; x *= 0x31848babu;
  x ^= x >> 14;
  return (float)(x >> 8) * (1.0f / 16777216.0f);
}

__global__ void dropout_relu_kernel(float* z, long long total, unsigned seed) {
  long long i = (long long)blockIdx.x * blockDim.x + threadIdx.x;
  if (i >= total) return;
  float v = z[i];
  float u = hash_uniform((unsigned)i ^ seed);
  v = (u >= DROP_P) ? v * (1.0f / (1.0f - DROP_P)) : 0.0f;
  z[i] = fmaxf(v, 0.0f);
}

// ------------------------- decode: dot of endpoint embeddings (C=64) --------
__global__ void decode_kernel(const float* __restrict__ Z,
                              const long long* __restrict__ ea,
                              const long long* __restrict__ eb,
                              float* __restrict__ out, int nEL) {
  int e = blockIdx.x * blockDim.x + threadIdx.x;
  if (e >= nEL) return;
  const float4* pa = (const float4*)(Z + (size_t)ea[e] * 64);
  const float4* pb = (const float4*)(Z + (size_t)eb[e] * 64);
  float s = 0.f;
#pragma unroll
  for (int i = 0; i < 16; ++i) {
    float4 a = pa[i], b = pb[i];
    s += a.x * b.x + a.y * b.y + a.z * b.z + a.w * b.w;
  }
  out[e] = s;
}

// ---------------------------------------------------------------------------
static inline size_t align256(size_t x) { return (x + 255) & ~(size_t)255; }

extern "C" void kernel_launch(void* const* d_in, const int* in_sizes, int n_in,
                              void* d_out, int out_size, void* d_ws, size_t ws_size,
                              hipStream_t stream) {
  const float*     x   = (const float*)d_in[0];
  const long long* ei  = (const long long*)d_in[1];   // int64 per reference
  const long long* eli = (const long long*)d_in[2];
  const float* W1 = (const float*)d_in[3]; const float* b1 = (const float*)d_in[4];
  const float* W2 = (const float*)d_in[5]; const float* b2 = (const float*)d_in[6];
  const float* W3 = (const float*)d_in[7]; const float* b3 = (const float*)d_in[8];
  float* out = (float*)d_out;

  const int       N  = in_sizes[0] / 128;            // 50000 nodes
  const long long E  = in_sizes[1] / 2;              // 800000 edges
  const int       EL = in_sizes[2] / 2;              // 100000 label edges
  const long long* src = ei;
  const long long* dst = ei + E;

  char* ws = (char*)d_ws;
  size_t off = 0;
  float* dinv = (float*)(ws + off); off += align256((size_t)N * 4);
  float* bufA = (float*)(ws + off); off += align256((size_t)N * 128 * 4);  // h
  float* bufB = (float*)(ws + off); off += align256((size_t)N * 128 * 4);  // agg/z
  float* bufC = (float*)(ws + off); off += align256((size_t)N * 128 * 4);  // agg/z
  (void)ws_size;

  const int B = 256;
  const int gN    = (N + B - 1) / B;
  const int gE    = (int)((E + B - 1) / B);
  const long long tot128 = (long long)N * 128;
  const long long tot64  = (long long)N * 64;
  const int g128  = (int)((tot128 + B - 1) / B);
  const int g64   = (int)((tot64 + B - 1) / B);
  const int gGemm = (N + 127) / 128;
  const long long sc128 = (E + N) << 5;   // float4 slots, C=128
  const long long sc64  = (E + N) << 4;   // float4 slots, C=64
  const int gSc128 = (int)((sc128 + B - 1) / B);
  const int gSc64  = (int)((sc64 + B - 1) / B);

  // ---- degrees -> dinv ----
  init_deg_kernel<<<gN, B, 0, stream>>>(dinv, N);
  accum_deg_kernel<<<gE, B, 0, stream>>>(dinv, dst, E);
  finish_dinv_kernel<<<gN, B, 0, stream>>>(dinv, N);

  // ---- layer 1: z1 = relu(dropout(scatter(x@W1) + b1)) ----
  gemm_f16_wmma_kernel<8><<<gGemm, B, 0, stream>>>(x, W1, bufA, N);
  init_bias_kernel<<<g128, B, 0, stream>>>(bufB, b1, 127, tot128);
  scatter_edges_kernel<<<gSc128, B, 0, stream>>>(bufA, bufB, src, dst, dinv, E, N, 5, 128);
  dropout_relu_kernel<<<g128, B, 0, stream>>>(bufB, tot128, 0xA341316Cu);

  // ---- layer 2 ----
  gemm_f16_wmma_kernel<8><<<gGemm, B, 0, stream>>>(bufB, W2, bufA, N);
  init_bias_kernel<<<g128, B, 0, stream>>>(bufC, b2, 127, tot128);
  scatter_edges_kernel<<<gSc128, B, 0, stream>>>(bufA, bufC, src, dst, dinv, E, N, 5, 128);
  dropout_relu_kernel<<<g128, B, 0, stream>>>(bufC, tot128, 0xC8013EA4u);

  // ---- layer 3 (no dropout/relu) ----
  gemm_f16_wmma_kernel<4><<<gGemm, B, 0, stream>>>(bufC, W3, bufA, N);
  init_bias_kernel<<<g64, B, 0, stream>>>(bufB, b3, 63, tot64);
  scatter_edges_kernel<<<gSc64, B, 0, stream>>>(bufA, bufB, src, dst, dinv, E, N, 4, 64);

  // ---- decode ----
  decode_kernel<<<(EL + B - 1) / B, B, 0, stream>>>(bufB, eli, eli + EL, out, EL);
}
